// GTM_attention_54623394071036
// MI455X (gfx1250) — matrix operand order repaired and checked by reading, not compiled, not run on hardware
//
#include <hip/hip_runtime.h>

typedef __attribute__((ext_vector_type(16))) _Float16 v16h;
typedef __attribute__((ext_vector_type(8)))  float    v8f;

#define NN   100000
#define EE   1600000
#define DIN  128
#define DATT 64
#define DOUT 128
#define NEG_SLOPE 0.2f
#define EPS_ 1e-15f

// ---------------------------------------------------------------------------
// Convert the three weight matrices (already row-major [out][in]) to f16.
// ---------------------------------------------------------------------------
__global__ void wconv_kernel(const float* __restrict__ Wrow,
                             const float* __restrict__ Wcol,
                             const float* __restrict__ Wx,
                             _Float16* __restrict__ hWrow,
                             _Float16* __restrict__ hWcol,
                             _Float16* __restrict__ hWx) {
    int i = blockIdx.x * blockDim.x + threadIdx.x;
    if (i < DATT * DIN) {
        hWrow[i] = (_Float16)Wrow[i];
        hWcol[i] = (_Float16)Wcol[i];
    }
    if (i < DOUT * DIN) hWx[i] = (_Float16)Wx[i];
}

// ---------------------------------------------------------------------------
// a_row = x @ W_row^T ; a_col = x @ W_col^T   via v_wmma_f32_16x16x32_f16.
// One block = one 16-node tile; 8 waves: waves 0-3 -> a_row ntiles 0-3,
// waves 4-7 -> a_col ntiles 0-3.  x tile staged in LDS as f32, converted
// to f16 fragments in contiguous k order (k = kh*16 + s), matching B.
// ---------------------------------------------------------------------------
__global__ __launch_bounds__(256) void gemm_att_kernel(
    const float* __restrict__ x,
    const _Float16* __restrict__ hWrow,
    const _Float16* __restrict__ hWcol,
    float* __restrict__ a_row, float* __restrict__ a_col) {
    __shared__ float xs[16 * DIN];
    const int tile = blockIdx.x;
    const int tid  = threadIdx.x;
    const float* xg = x + (size_t)tile * 16 * DIN;
#pragma unroll
    for (int i = 0; i < 8; ++i) xs[tid + i * 256] = xg[tid + i * 256];
    __syncthreads();

    const int wave  = tid >> 5;
    const int lane  = tid & 31;
    const bool isrow = (wave < 4);
    const int ntile = isrow ? wave : wave - 4;
    const _Float16* hW = isrow ? hWrow : hWcol;
    float* outp        = isrow ? a_row : a_col;

    const int m  = lane & 15;   // A row / B column within tile
    const int kh = lane >> 4;   // lane-half selects K sub-range

    v8f c = {};
#pragma unroll
    for (int kb = 0; kb < DIN; kb += 32) {
        v16h a, b;
        const float* xr = &xs[m * DIN + kb + kh * 16];
#pragma unroll
        for (int j = 0; j < 8; ++j) {
            a[j]     = (_Float16)xr[j];        // VGPR0-3: K = kh*16 + 0..7
            a[8 + j] = (_Float16)xr[8 + j];    // VGPR4-7: K = kh*16 + 8..15
        }
        // B: 16 consecutive halfs of W[n][k], K = kh*16 + 0..15 (32B aligned)
        b = *(const v16h*)(hW + (size_t)(ntile * 16 + m) * DIN + kb + kh * 16);
        c = __builtin_amdgcn_wmma_f32_16x16x32_f16(false, a, false, b,
                                                   (short)0, c, false, false);
    }
#pragma unroll
    for (int r = 0; r < 8; ++r) {
        int mm = kh * 8 + r;  // C layout: VGPR r -> M=r (lanes<16) / M=8+r
        outp[((size_t)tile * 16 + mm) * DATT + ntile * 16 + m] = c[r];
    }
}

// ---------------------------------------------------------------------------
// row_ptr[i] = lower_bound(row, i)  (row is sorted) -> atomic-free segments.
// ---------------------------------------------------------------------------
__global__ void rowptr_kernel(const int* __restrict__ row, int* __restrict__ ptr) {
    int i = blockIdx.x * blockDim.x + threadIdx.x;
    if (i > NN) return;
    int lo = 0, hi = EE;
    while (lo < hi) {
        int mid = (lo + hi) >> 1;
        if (row[mid] < i) lo = mid + 1; else hi = mid;
    }
    ptr[i] = lo;
}

// ---------------------------------------------------------------------------
// Per-edge score: s = leaky_relu(dot(a_row[row], a_col[col]) * rsqrt(DATT)).
// Block-reduced max -> order-independent atomicMax on monotonic uint key.
// ---------------------------------------------------------------------------
__global__ __launch_bounds__(256) void score_kernel(
    const float* __restrict__ a_row, const float* __restrict__ a_col,
    const int* __restrict__ row, const int* __restrict__ col,
    float* __restrict__ s, unsigned* __restrict__ smax_u) {
    int e = blockIdx.x * 256 + threadIdx.x;
    float val = -1e30f;
    if (e < EE) {
        const float4* ar = (const float4*)(a_row + (size_t)row[e] * DATT);
        const float4* ac = (const float4*)(a_col + (size_t)col[e] * DATT);
        float acc = 0.f;
#pragma unroll
        for (int j = 0; j < DATT / 4; ++j) {
            float4 p = ar[j], q = ac[j];
            acc += p.x * q.x + p.y * q.y + p.z * q.z + p.w * q.w;
        }
        acc *= 0.125f;  // rsqrt(64)
        val = (acc >= 0.f) ? acc : NEG_SLOPE * acc;
        s[e] = val;
    }
    __shared__ float red[256];
    red[threadIdx.x] = val;
    __syncthreads();
    for (int off = 128; off; off >>= 1) {
        if (threadIdx.x < off)
            red[threadIdx.x] = fmaxf(red[threadIdx.x], red[threadIdx.x + off]);
        __syncthreads();
    }
    if (threadIdx.x == 0) {
        unsigned k = __float_as_uint(red[0]);
        k = (k & 0x80000000u) ? ~k : (k | 0x80000000u);  // monotonic map
        atomicMax(smax_u, k);
    }
}

// ---------------------------------------------------------------------------
// Per-node: atten = exp(s - smax) (written in place over s);
// deg = sum(atten); deg2 = (sum mask*atten)/(deg+eps) [constant per segment];
// sc[r] = 1/(deg+eps) * 1/(deg2+eps)  -> final edge weight
//         w_e = mask_e * atten_e * sc[row_e].
// ---------------------------------------------------------------------------
__global__ void deg_kernel(float* __restrict__ s, const float* __restrict__ mask,
                           const int* __restrict__ ptr,
                           const unsigned* __restrict__ smax_u,
                           float* __restrict__ sc) {
    int r = blockIdx.x * blockDim.x + threadIdx.x;
    if (r >= NN) return;
    unsigned k = *smax_u;
    float smax = (k & 0x80000000u) ? __uint_as_float(k & 0x7FFFFFFFu)
                                   : __uint_as_float(~k);
    int e0 = ptr[r], e1 = ptr[r + 1];
    float deg = 0.f, m2 = 0.f;
    for (int e = e0; e < e1; ++e) {
        float a = __expf(s[e] - smax);
        s[e] = a;                 // reuse buffer as atten
        deg += a;
        m2  += mask[e] * a;
    }
    float s1 = 1.0f / (deg + EPS_);
    float d2 = m2 * s1;
    float s2 = 1.0f / (d2 + EPS_);
    sc[r] = s1 * s2;
}

// ---------------------------------------------------------------------------
// Aggregation: one wave per node, lane holds 4 of 128 features.
// agg[r] = sum_e w_e * x[col_e]   (atomic-free; coalesced 512B x-gathers).
// ---------------------------------------------------------------------------
__global__ __launch_bounds__(256) void agg_kernel(
    const float* __restrict__ x, const float* __restrict__ mask,
    const float* __restrict__ atten, const int* __restrict__ col,
    const int* __restrict__ ptr, const float* __restrict__ sc,
    float* __restrict__ agg) {
    int node = blockIdx.x * 8 + (threadIdx.x >> 5);
    int lane = threadIdx.x & 31;
    if (node >= NN) return;
    int e0 = ptr[node], e1 = ptr[node + 1];
    float cscale = sc[node];
    float4 acc = {0.f, 0.f, 0.f, 0.f};
    for (int e = e0; e < e1; ++e) {
        float w = mask[e] * atten[e] * cscale;
        float4 xv = ((const float4*)(x + (size_t)col[e] * DIN))[lane];
        acc.x += w * xv.x; acc.y += w * xv.y;
        acc.z += w * xv.z; acc.w += w * xv.w;
    }
    ((float4*)(agg + (size_t)node * DIN))[lane] = acc;
}

// ---------------------------------------------------------------------------
// out = agg @ W_x^T + b.  Same WMMA scheme; agg aliases d_out, which is safe
// because each block stages its own 16x128 tile into LDS before overwriting.
// ---------------------------------------------------------------------------
__global__ __launch_bounds__(256) void gemm_out_kernel(
    const float* __restrict__ agg, const _Float16* __restrict__ hWx,
    const float* __restrict__ bias, float* __restrict__ out) {
    __shared__ float xs[16 * DIN];
    const int tile = blockIdx.x;
    const int tid  = threadIdx.x;
    const float* xg = agg + (size_t)tile * 16 * DIN;
#pragma unroll
    for (int i = 0; i < 8; ++i) xs[tid + i * 256] = xg[tid + i * 256];
    __syncthreads();

    const int wave = tid >> 5;   // ntile 0..7 (DOUT=128)
    const int lane = tid & 31;
    const int m  = lane & 15;
    const int kh = lane >> 4;

    v8f c = {};
#pragma unroll
    for (int kb = 0; kb < DIN; kb += 32) {
        v16h a, b;
        const float* xr = &xs[m * DIN + kb + kh * 16];
#pragma unroll
        for (int j = 0; j < 8; ++j) {
            a[j]     = (_Float16)xr[j];
            a[8 + j] = (_Float16)xr[8 + j];
        }
        b = *(const v16h*)(hWx + (size_t)(wave * 16 + m) * DIN + kb + kh * 16);
        c = __builtin_amdgcn_wmma_f32_16x16x32_f16(false, a, false, b,
                                                   (short)0, c, false, false);
    }
    float bv = bias[wave * 16 + m];
#pragma unroll
    for (int r = 0; r < 8; ++r) {
        int mm = kh * 8 + r;
        out[((size_t)tile * 16 + mm) * DOUT + wave * 16 + m] = c[r] + bv;
    }
}

// ---------------------------------------------------------------------------
extern "C" void kernel_launch(void* const* d_in, const int* in_sizes, int n_in,
                              void* d_out, int out_size, void* d_ws, size_t ws_size,
                              hipStream_t stream) {
    const float* x     = (const float*)d_in[0];
    const float* maskv = (const float*)d_in[1];
    const float* Wrow  = (const float*)d_in[2];
    const float* Wcol  = (const float*)d_in[3];
    const float* Wx    = (const float*)d_in[4];
    const float* bx    = (const float*)d_in[5];
    const int*   row   = (const int*)d_in[6];
    const int*   col   = (const int*)d_in[7];
    float* out = (float*)d_out;

    // workspace carve-up (256B aligned), ~58.5 MB total
    char* p = (char*)d_ws;
    auto carve = [&](size_t bytes) -> char* {
        char* r = p;
        p += (bytes + 255) & ~(size_t)255;
        return r;
    };
    float*    a_row = (float*)carve((size_t)NN * DATT * 4);
    float*    a_col = (float*)carve((size_t)NN * DATT * 4);
    float*    s     = (float*)carve((size_t)EE * 4);       // scores -> atten
    float*    sc    = (float*)carve((size_t)NN * 4);
    int*      ptr   = (int*)  carve((size_t)(NN + 1) * 4);
    unsigned* smax  = (unsigned*)carve(256);
    _Float16* hWrow = (_Float16*)carve((size_t)DATT * DIN * 2);
    _Float16* hWcol = (_Float16*)carve((size_t)DATT * DIN * 2);
    _Float16* hWx   = (_Float16*)carve((size_t)DOUT * DIN * 2);
    float*    agg   = out;  // aliased: rewritten tile-locally by gemm_out

    hipMemsetAsync(smax, 0, 4, stream);

    wconv_kernel<<<(DOUT * DIN + 255) / 256, 256, 0, stream>>>(
        Wrow, Wcol, Wx, hWrow, hWcol, hWx);

    gemm_att_kernel<<<NN / 16, 256, 0, stream>>>(x, hWrow, hWcol, a_row, a_col);

    rowptr_kernel<<<(NN + 1 + 255) / 256, 256, 0, stream>>>(row, ptr);

    score_kernel<<<EE / 256, 256, 0, stream>>>(a_row, a_col, row, col, s, smax);

    deg_kernel<<<(NN + 255) / 256, 256, 0, stream>>>(s, maskv, ptr, smax, sc);

    agg_kernel<<<NN / 8, 256, 0, stream>>>(x, maskv, s, col, ptr, sc, agg);

    gemm_out_kernel<<<NN / 16, 256, 0, stream>>>(agg, hWx, bx, out);
}
